// Recurrence_64115271795051
// MI455X (gfx1250) — matrix operand order repaired and checked by reading, not compiled, not run on hardware
//
#include <hip/hip_runtime.h>
#include <hip/hip_bf16.h>
#include <math.h>

// ---------------------------------------------------------------------------
// Problem constants (from the reference)
// ---------------------------------------------------------------------------
constexpr int kH  = 128;   // hidden
constexpr int kHW = 256;   // 16x16 pixels
constexpr int kN  = 128;   // batch
constexpr int kS  = 642;   // 1 + 1 + 128 + 512
constexpr int kG  = 384;   // 3*H (GRU gates)

// ---------------------------------------------------------------------------
// WMMA helpers (CDNA5 / gfx1250, wave32).  D = A(16x32 f16) x B(32x16 f16) + C
//   A: lane L -> row m = L&15;  group g (VGPR 4g..4g+3): K = g*16 + (L>=16?8:0) + 2w..
//   B: lane L -> col n = L&15;  same K order as A: group g: K = g*16 + (L>=16?8:0) + 2w..
//   C/D: lane L -> col n = L&15; VGPR r: row m = (L>=16?8:0) + r
// ---------------------------------------------------------------------------
typedef __attribute__((ext_vector_type(16))) _Float16 v16h;
typedef __attribute__((ext_vector_type(8)))  float    v8f;

union AFrag { v16h v; unsigned int u[8]; };

__device__ __forceinline__ unsigned short f2h(float x) {
  union { _Float16 h; unsigned short s; } c;
  c.h = (_Float16)x;
  return c.s;
}

__device__ __forceinline__ void loadA32(const unsigned short* rowp, int hi, AFrag& f) {
  const unsigned short* q = rowp + hi * 8;
#pragma unroll
  for (int w = 0; w < 4; ++w) f.u[w]     = *(const unsigned int*)(q + 2 * w);
#pragma unroll
  for (int w = 0; w < 4; ++w) f.u[4 + w] = *(const unsigned int*)(q + 16 + 2 * w);
}

// Two independent 16-half K blocks (conv1: one 3x3 position = 16 channels)
__device__ __forceinline__ void loadApairBoth(const unsigned short* pa,
                                              const unsigned short* pb, int hi,
                                              AFrag& f) {
  const unsigned short* qa = pa + hi * 8;
  const unsigned short* qb = pb + hi * 8;
#pragma unroll
  for (int w = 0; w < 4; ++w) f.u[w]     = *(const unsigned int*)(qa + 2 * w);
#pragma unroll
  for (int w = 0; w < 4; ++w) f.u[4 + w] = *(const unsigned int*)(qb + 2 * w);
}

__device__ __forceinline__ void loadApairLow(const unsigned short* pa, int hi,
                                             AFrag& f) {
  const unsigned short* qa = pa + hi * 8;
#pragma unroll
  for (int w = 0; w < 4; ++w) f.u[w] = *(const unsigned int*)(qa + 2 * w);
#pragma unroll
  for (int w = 0; w < 4; ++w) f.u[4 + w] = 0u;
}

__device__ __forceinline__ void loadB32(const unsigned short* rowp, int hi, AFrag& f) {
  const unsigned short* q = rowp + hi * 8;
#pragma unroll
  for (int w = 0; w < 4; ++w) f.u[w]     = *(const unsigned int*)(q + 2 * w);
#pragma unroll
  for (int w = 0; w < 4; ++w) f.u[4 + w] = *(const unsigned int*)(q + 16 + 2 * w);
}

__device__ __forceinline__ void wmma_f16(v8f& c, const AFrag& a, const AFrag& b) {
  c = __builtin_amdgcn_wmma_f32_16x16x32_f16(false, a.v, false, b.v, (short)0, c,
                                             false, false);
}

__device__ __forceinline__ float sigmoidf_(float x) { return 1.f / (1.f + expf(-x)); }

// ---- CDNA5 async copy: global -> LDS, tracked with ASYNCcnt -----------------
__device__ __forceinline__ unsigned lds_off_u32(const void* p) {
  // generic pointers to LDS carry the LDS byte offset in the low 32 bits
  return (unsigned)(unsigned long long)p;
}
__device__ __forceinline__ void async_load_b128(unsigned lds_byte, const void* gptr) {
  asm volatile("global_load_async_to_lds_b128 %0, %1, off"
               :: "v"(lds_byte), "v"((unsigned long long)gptr)
               : "memory");
}
#define ASYNC_WAIT_6() asm volatile("s_wait_asynccnt 0x6" ::: "memory")
#define ASYNC_WAIT_0() asm volatile("s_wait_asynccnt 0x0" ::: "memory")

// ---------------------------------------------------------------------------
// Kernel 0: convert weights f32 -> f16 (+ conv-weight reorder for implicit GEMM)
// ---------------------------------------------------------------------------
__global__ void __launch_bounds__(256) k_prep(
    const float* wihF, const float* whhF, const float* wihB, const float* whhB,
    const float* cWih, const float* cWhh, const float* qgW,
    const float* c1w, const float* c2w,
    unsigned short* dWihF, unsigned short* dWhhF, unsigned short* dWihB,
    unsigned short* dWhhB, unsigned short* dCWih, unsigned short* dCWhh,
    unsigned short* dQgW, unsigned short* dW1, unsigned short* dW2) {
  int task = blockIdx.y;
  int idx  = blockIdx.x * 256 + threadIdx.x;
  if (task < 7) {
    const float* srcs[7]    = {wihF, whhF, wihB, whhB, cWih, cWhh, qgW};
    unsigned short* dsts[7] = {dWihF, dWhhF, dWihB, dWhhB, dCWih, dCWhh, dQgW};
    int sizes[7] = {49152, 49152, 49152, 49152, 49152, 49152, 32768};
    if (idx < sizes[task]) dsts[task][idx] = f2h(srcs[task][idx]);
  } else if (task == 7) {
    // conv1_w (128,16,3,3) -> (128 x 160): K = pos*16 + d, zero-padded 144->160
    if (idx < 128 * 160) {
      int c = idx / 160, k = idx % 160;
      float v = 0.f;
      if (k < 144) { int pos = k >> 4, d = k & 15; v = c1w[c * 144 + d * 9 + pos]; }
      dW1[idx] = f2h(v);
    }
  } else {
    // conv2_w (128,128,3,3) -> (128 x 1152): K = pos*128 + ci
    if (idx < 128 * 1152) {
      int co = idx / 1152, k = idx % 1152;
      int pos = k >> 7, ci = k & 127;
      dW2[idx] = f2h(c2w[co * 1152 + ci * 9 + pos]);
    }
  }
}

// ---------------------------------------------------------------------------
// Kernel 1: fused conv1+conv2 (implicit-GEMM WMMA), one workgroup per batch n.
// LDS images carry an 18x18 zero halo -> branchless A-fragment loads.
// Produces Mh[n][p][c] (f16).
// ---------------------------------------------------------------------------
__global__ void __launch_bounds__(256) k_conv(
    const float* __restrict__ inputs, const float* __restrict__ b1,
    const float* __restrict__ b2, const unsigned short* __restrict__ w1,
    const unsigned short* __restrict__ w2, unsigned short* __restrict__ Mh) {
  __shared__ __align__(16) unsigned short obs[324 * 16];   // 18x18 halo, [py*18+px][d]
  __shared__ __align__(16) unsigned short Y1[324 * 136];   // conv1 out, halo, pad 136
  const int n = blockIdx.x;
  const int tid = threadIdx.x;
  const int lane = tid & 31, wave = tid >> 5;
  const int ln = lane & 15, hi = lane >> 4;
  const float* in0 = inputs + (size_t)n * 4097;

  {  // zero both LDS images (covers the halo)
    unsigned* z0 = (unsigned*)obs;
    for (int i = tid; i < 324 * 16 / 2; i += 256) z0[i] = 0u;
    unsigned* z1 = (unsigned*)Y1;
    for (int i = tid; i < 324 * 136 / 2; i += 256) z1[i] = 0u;
  }
  __syncthreads();
  for (int i = tid; i < 4096; i += 256) {
    int p = i >> 4, d = i & 15, y = p >> 4, x = p & 15;
    obs[((y + 1) * 18 + (x + 1)) * 16 + d] = f2h(in0[i]);
  }
  __syncthreads();

  // conv1: K = 9 positions * 16 ch (two positions per WMMA K-window)
  for (int tile = wave; tile < 128; tile += 8) {
    int mt = tile >> 3, nt = tile & 7;
    int p = mt * 16 + ln, y = p >> 4, x = p & 15;
    v8f acc = {0.f, 0.f, 0.f, 0.f, 0.f, 0.f, 0.f, 0.f};
#pragma unroll
    for (int w = 0; w < 5; ++w) {
      int posA = 2 * w, posB = 2 * w + 1;
      const unsigned short* pa = &obs[((y + posA / 3) * 18 + (x + posA % 3)) * 16];
      AFrag a, b;
      if (posB < 9) {
        const unsigned short* pb = &obs[((y + posB / 3) * 18 + (x + posB % 3)) * 16];
        loadApairBoth(pa, pb, hi, a);
      } else {
        loadApairLow(pa, hi, a);
      }
      loadB32(w1 + (nt * 16 + ln) * 160 + w * 32, hi, b);
      wmma_f16(acc, a, b);
    }
    int c = nt * 16 + ln;
    float bias = b1[c];
#pragma unroll
    for (int r = 0; r < 8; ++r) {
      int pr = mt * 16 + hi * 8 + r;
      float v = acc[r] + bias;
      Y1[(((pr >> 4) + 1) * 18 + (pr & 15) + 1) * 136 + c] = f2h(v > 0.f ? v : 0.f);
    }
  }
  __syncthreads();

  // conv2: K = 9 positions * 128 ch, branchless A loads via halo
  for (int tile = wave; tile < 128; tile += 8) {
    int mt = tile >> 3, nt = tile & 7;
    int p = mt * 16 + ln, y = p >> 4, x = p & 15;
    v8f acc = {0.f, 0.f, 0.f, 0.f, 0.f, 0.f, 0.f, 0.f};
#pragma unroll
    for (int pos = 0; pos < 9; ++pos) {
      const unsigned short* rowp =
          &Y1[((y + pos / 3) * 18 + (x + pos % 3)) * 136];
#pragma unroll
      for (int kk = 0; kk < 4; ++kk) {
        AFrag a, b;
        loadA32(rowp + kk * 32, hi, a);
        loadB32(w2 + (size_t)(nt * 16 + ln) * 1152 + pos * 128 + kk * 32, hi, b);
        wmma_f16(acc, a, b);
      }
    }
    int c = nt * 16 + ln;
    float bias = b2[c];
#pragma unroll
    for (int r = 0; r < 8; ++r) {
      int pr = mt * 16 + hi * 8 + r;
      float v = acc[r] + bias;
      Mh[(size_t)n * kHW * kH + (size_t)pr * kH + c] = f2h(v > 0.f ? v : 0.f);
    }
  }
}

// ---------------------------------------------------------------------------
// Kernel 2: parallel GRU input projections gi = x@wih.T + bih, all 256
// positions, both directions.  B fragments loaded once per nt, swept over mt.
// ---------------------------------------------------------------------------
__global__ void __launch_bounds__(256) k_gi(
    const unsigned short* __restrict__ Mh, const unsigned short* __restrict__ wihF,
    const unsigned short* __restrict__ wihB, const float* __restrict__ bihF,
    const float* __restrict__ bihB, float* __restrict__ giF, float* __restrict__ giB) {
  const int p = blockIdx.x, dir = blockIdx.y;
  const unsigned short* W = dir ? wihB : wihF;
  const float* bih = dir ? bihB : bihF;
  float* gi = dir ? giB : giF;
  const int tid = threadIdx.x, lane = tid & 31, wave = tid >> 5;
  const int ln = lane & 15, hi = lane >> 4;

  for (int g3 = 0; g3 < 3; ++g3) {
    int nt = wave * 3 + g3;             // 8 waves * 3 = 24 gate tiles
    AFrag b[4];
#pragma unroll
    for (int kk = 0; kk < 4; ++kk)
      loadB32(W + (nt * 16 + ln) * kH + kk * 32, hi, b[kk]);
    int g = nt * 16 + ln;
    float bb = bih[g];
    for (int mt = 0; mt < 8; ++mt) {
      const unsigned short* arow = Mh + ((size_t)(mt * 16 + ln) * kHW + p) * kH;
      v8f acc = {0.f, 0.f, 0.f, 0.f, 0.f, 0.f, 0.f, 0.f};
#pragma unroll
      for (int kk = 0; kk < 4; ++kk) {
        AFrag a;
        loadA32(arow + kk * 32, hi, a);
        wmma_f16(acc, a, b[kk]);
      }
#pragma unroll
      for (int r = 0; r < 8; ++r) {
        int nn = mt * 16 + hi * 8 + r;
        gi[((size_t)p * kN + nn) * kG + g] = acc[r] + bb;
      }
    }
  }
}

// ---------------------------------------------------------------------------
// Kernel 3: sequential encoder scans.  2 dirs x 8 batch-groups of 16 rows =
// 16 independent workgroups (no inter-WG sync).  Per step: h@whh.T via WMMA
// with register-resident B fragments; gi streamed into a double-buffered LDS
// region with global_load_async_to_lds_b128 (ASYNCcnt-fenced).
// ---------------------------------------------------------------------------
__global__ void __launch_bounds__(256) k_encscan(
    const float* __restrict__ giF, const float* __restrict__ giB,
    const unsigned short* __restrict__ whhF, const unsigned short* __restrict__ whhB,
    const float* __restrict__ bhhF, const float* __restrict__ bhhB,
    float* __restrict__ K) {
  __shared__ __align__(16) unsigned short hb[16 * 136];  // h f16 (A-frags)
  __shared__ float hf[16 * 128];                         // h f32 carry
  __shared__ float gh[16 * 388];                         // h@whh.T staging
  __shared__ __align__(16) float gib[2 * 16 * 384];      // gi double buffer
  __shared__ float bhhL[384];
  const int dir = blockIdx.x >> 3, bg = blockIdx.x & 7;
  const float* gi = dir ? giB : giF;
  const unsigned short* Whh = dir ? whhB : whhF;
  const float* bhh = dir ? bhhB : bhhF;
  const int tid = threadIdx.x, lane = tid & 31, wave = tid >> 5;
  const int ln = lane & 15, hi = lane >> 4;

  for (int e = tid; e < 2048; e += 256) {
    hf[(e >> 7) * 128 + (e & 127)] = 0.f;
    hb[(e >> 7) * 136 + (e & 127)] = 0;
  }
  for (int e = tid; e < 384; e += 256) bhhL[e] = bhh[e];

  // Hoist this wave's 12 invariant B fragments (whh) into registers.
  AFrag Bw[3][4];
#pragma unroll
  for (int j = 0; j < 3; ++j)
#pragma unroll
    for (int kk = 0; kk < 4; ++kk)
      loadB32(Whh + ((wave + 8 * j) * 16 + ln) * kH + kk * 32, hi, Bw[j][kk]);

  // Preload gi for the first step into buffer 0.
  {
    int p0 = dir ? 255 : 0;
    const float* src = gi + ((size_t)p0 * kN + bg * 16) * kG;
    unsigned ldsbase = lds_off_u32(&gib[0]);
#pragma unroll
    for (int i = 0; i < 6; ++i) {
      unsigned e4 = (unsigned)(i * 256 + tid) * 4u;   // float index, 4 floats/lane
      async_load_b128(ldsbase + e4 * 4u, src + e4);
    }
  }
  __syncthreads();

  for (int s = 0; s < 256; ++s) {
    int p = dir ? 255 - s : s;
    int cur = s & 1, nxt = cur ^ 1;
    // Kick off async copy of next step's gi (clamped at the ends).
    {
      int pn = dir ? p - 1 : p + 1;
      pn = pn < 0 ? 0 : (pn > 255 ? 255 : pn);
      const float* src = gi + ((size_t)pn * kN + bg * 16) * kG;
      unsigned ldsbase = lds_off_u32(&gib[nxt * 6144]);
#pragma unroll
      for (int i = 0; i < 6; ++i) {
        unsigned e4 = (unsigned)(i * 256 + tid) * 4u;
        async_load_b128(ldsbase + e4 * 4u, src + e4);
      }
    }
    // gh = h @ whh.T   (M=16, N=384, K=128): 24 tiles, 3 per wave
#pragma unroll
    for (int j = 0; j < 3; ++j) {
      int nt = wave + 8 * j;
      v8f acc = {0.f, 0.f, 0.f, 0.f, 0.f, 0.f, 0.f, 0.f};
#pragma unroll
      for (int kk = 0; kk < 4; ++kk) {
        AFrag a;
        loadA32(&hb[ln * 136 + kk * 32], hi, a);
        wmma_f16(acc, a, Bw[j][kk]);
      }
      int g = nt * 16 + ln;
#pragma unroll
      for (int r = 0; r < 8; ++r) gh[(hi * 8 + r) * 388 + g] = acc[r];
    }
    ASYNC_WAIT_6();   // current buffer resident (next's 6 may stay outstanding)
    __syncthreads();
    // gates + state update + K store
    const float* gcur = &gib[cur * 6144];
    for (int e = tid; e < 2048; e += 256) {
      int nn = e >> 7, c = e & 127;
      const float* gr = gcur + nn * kG;
      float ir = gr[c], iz = gr[128 + c], inn = gr[256 + c];
      float hr = gh[nn * 388 + c] + bhhL[c];
      float hz = gh[nn * 388 + 128 + c] + bhhL[128 + c];
      float hn = gh[nn * 388 + 256 + c] + bhhL[256 + c];
      float r_ = sigmoidf_(ir + hr);
      float z_ = sigmoidf_(iz + hz);
      float nv = tanhf(inn + r_ * hn);
      float hnew = (1.f - z_) * nv + z_ * hf[nn * 128 + c];
      hf[nn * 128 + c] = hnew;
      hb[nn * 136 + c] = f2h(hnew);
      int nabs = bg * 16 + nn;
      K[((size_t)nabs * kHW + p) * 256 + dir * 128 + c] = hnew;
    }
    __syncthreads();
  }
  ASYNC_WAIT_0();
}

// ---------------------------------------------------------------------------
// Kernel 4: decoder scan (64 steps).  Batch-parallel: 8 WGs x 16 batch rows.
// ---------------------------------------------------------------------------
__global__ void __launch_bounds__(256) k_dec(
    const unsigned short* __restrict__ Mh, const float* __restrict__ K,
    const unsigned short* __restrict__ qgW, const unsigned short* __restrict__ cWih,
    const unsigned short* __restrict__ cWhh, const float* __restrict__ qg_b,
    const float* __restrict__ term_w, const float* __restrict__ term_b,
    const float* __restrict__ critic_w, const float* __restrict__ critic_b,
    const float* __restrict__ cell_bih, const float* __restrict__ cell_bhh,
    const float* __restrict__ inputs, const float* __restrict__ gumbel,
    const float* __restrict__ rnn_hxs, float* __restrict__ out) {
  __shared__ __align__(16) unsigned short hb[16 * 136];
  __shared__ float hf[16 * 128];
  __shared__ __align__(16) unsigned short rb[16 * 136];
  __shared__ __align__(16) float kv[16 * 256];
  __shared__ float wv[16 * 256];
  __shared__ float gib[16 * 388];
  __shared__ float ghb[16 * 388];
  __shared__ float dnn[16];
  __shared__ int asel[16];
  __shared__ float cbihL[384], cbhhL[384], qgbL[256], twL[128], cwL[128];
  const int n0 = blockIdx.x * 16;
  const int tid = threadIdx.x, lane = tid & 31, wave = tid >> 5;
  const int ln = lane & 15, hi = lane >> 4;
  const float tb0 = term_b[0], cb0 = critic_b[0];

  for (int e = tid; e < 2048; e += 256) {
    int nn = e >> 7, c = e & 127;
    float v = rnn_hxs[(size_t)(n0 + nn) * kS + 2 + c];
    hf[nn * 128 + c] = v;
    hb[nn * 136 + c] = f2h(v);
  }
  for (int e = tid; e < 384; e += 256) { cbihL[e] = cell_bih[e]; cbhhL[e] = cell_bhh[e]; }
  if (tid < 256) qgbL[tid] = qg_b[tid];
  if (tid < 128) { twL[tid] = term_w[tid]; cwL[tid] = critic_w[tid]; }
  __syncthreads();

  for (int t = 0; t < 64; ++t) {
    // 1) k = h @ qg_w.T + qg_b  (M=16, N=256, K=128): 16 tiles, 2 per wave
#pragma unroll
    for (int j = 0; j < 2; ++j) {
      int nt = wave + 8 * j;
      v8f acc = {0.f, 0.f, 0.f, 0.f, 0.f, 0.f, 0.f, 0.f};
#pragma unroll
      for (int kk = 0; kk < 4; ++kk) {
        AFrag a, b;
        loadA32(&hb[ln * 136 + kk * 32], hi, a);
        loadB32(qgW + (nt * 16 + ln) * kH + kk * 32, hi, b);
        wmma_f16(acc, a, b);
      }
      int g = nt * 16 + ln;
      float bb = qgbL[g];
#pragma unroll
      for (int r = 0; r < 8; ++r) kv[(hi * 8 + r) * 256 + g] = acc[r] + bb;
    }
    // 2) done = sigmoid(h . term_w + term_b)
    if (tid < 16) {
      float s = 0.f;
      for (int c = 0; c < 128; ++c) s += hf[tid * 128 + c] * twL[c];
      dnn[tid] = sigmoidf_(s + tb0);
    }
    __syncthreads();
    // 3) w[n][l] = K[n][l][:] . k[n][:]
    {
      int nn = tid >> 4, sl = tid & 15, nabs = n0 + nn;
      const float4* kk4 = (const float4*)&kv[nn * 256];
      if (t < 63) {  // prefetch next step's gumbel rows
        const char* pf =
            (const char*)(gumbel + ((size_t)(t + 1) * kN + nabs) * 512 + sl * 32);
        __builtin_prefetch(pf, 0, 1);
      }
      for (int j = 0; j < 16; ++j) {
        int l = sl + 16 * j;
        const float4* kr = (const float4*)(K + ((size_t)nabs * kHW + l) * 256);
        float s = 0.f;
        for (int q = 0; q < 64; ++q) {
          float4 a = kr[q], b = kk4[q];
          s += a.x * b.x + a.y * b.y + a.z * b.z + a.w * b.w;
        }
        wv[nn * 256 + l] = s;
      }
    }
    __syncthreads();
    // 4) logits / softmax / gumbel-argmax / probs output
    {
      int nn = tid >> 4, sl = tid & 15, nabs = n0 + nn;
      float done = dnn[nn];
      const float* gmb = gumbel + ((size_t)t * kN + nabs) * 512;
      float lg[32];
      float mx = -3.4e38f, pbest = -3.4e38f;
      int ibest = 0x7fffffff;
#pragma unroll
      for (int i = 0; i < 32; ++i) {
        int jj = sl + 16 * i;
        float lv = (jj < 256) ? (1.f - done) * wv[nn * 256 + jj]
                              : done * wv[nn * 256 + jj - 256];
        lg[i] = lv;
        mx = fmaxf(mx, lv);
        float pert = lv + gmb[jj];
        if (pert > pbest || (pert == pbest && jj < ibest)) { pbest = pert; ibest = jj; }
      }
      for (int m = 8; m >= 1; m >>= 1) {  // 16-lane group reduce (wave32)
        mx = fmaxf(mx, __shfl_xor(mx, m, 32));
        float ov = __shfl_xor(pbest, m, 32);
        int oi = __shfl_xor(ibest, m, 32);
        if (ov > pbest || (ov == pbest && oi < ibest)) { pbest = ov; ibest = oi; }
      }
      float se = 0.f;
#pragma unroll
      for (int i = 0; i < 32; ++i) se += expf(lg[i] - mx);
      for (int m = 8; m >= 1; m >>= 1) se += __shfl_xor(se, m, 32);
      float apf = inputs[((size_t)t * kN + nabs) * 4097 + 4096];
      int ap = (int)apf;
      int a = (ap < 0) ? ibest : ap;
      if (sl == 0) asel[nn] = a;
      float* ob = out + ((size_t)t * kN + nabs) * kS;
      float* obT = out + ((size_t)(64 * kN) + nabs) * kS;
      float inv = 1.f / se;
#pragma unroll
      for (int i = 0; i < 32; ++i) {
        int jj = sl + 16 * i;
        float pr = expf(lg[i] - mx) * inv;
        ob[130 + jj] = pr;
        if (t == 63) obT[130 + jj] = pr;
      }
      if (sl == 0) { ob[0] = (float)a; if (t == 63) obT[0] = (float)a; }
    }
    __syncthreads();
    // 5) gather r = M[n, a%256, :]  (f16)
    for (int e = tid; e < 2048; e += 256) {
      int nn = e >> 7, c = e & 127, a = asel[nn] & 255;
      rb[nn * 136 + c] = Mh[((size_t)(n0 + nn) * kHW + a) * kH + c];
    }
    __syncthreads();
    // 6) gi = r@cell_wih.T, gh = h@cell_whh.T : 48 tile jobs, 6 per wave
#pragma unroll
    for (int j = 0; j < 6; ++j) {
      int job = wave + 8 * j;
      bool isGi = job < 24;
      int nt = isGi ? job : job - 24;
      const unsigned short* A = isGi ? rb : hb;
      const unsigned short* W = isGi ? cWih : cWhh;
      float* G = isGi ? gib : ghb;
      v8f acc = {0.f, 0.f, 0.f, 0.f, 0.f, 0.f, 0.f, 0.f};
#pragma unroll
      for (int kk = 0; kk < 4; ++kk) {
        AFrag a, b;
        loadA32(A + ln * 136 + kk * 32, hi, a);
        loadB32(W + (nt * 16 + ln) * kH + kk * 32, hi, b);
        wmma_f16(acc, a, b);
      }
      int g = nt * 16 + ln;
#pragma unroll
      for (int r = 0; r < 8; ++r) G[(hi * 8 + r) * 388 + g] = acc[r];
    }
    __syncthreads();
    // 7) GRU gates -> h2
    for (int e = tid; e < 2048; e += 256) {
      int nn = e >> 7, c = e & 127;
      float ir = gib[nn * 388 + c] + cbihL[c];
      float iz = gib[nn * 388 + 128 + c] + cbihL[128 + c];
      float inn = gib[nn * 388 + 256 + c] + cbihL[256 + c];
      float hr = ghb[nn * 388 + c] + cbhhL[c];
      float hz = ghb[nn * 388 + 128 + c] + cbhhL[128 + c];
      float hn = ghb[nn * 388 + 256 + c] + cbhhL[256 + c];
      float r_ = sigmoidf_(ir + hr);
      float z_ = sigmoidf_(iz + hz);
      float nv = tanhf(inn + r_ * hn);
      float hnew = (1.f - z_) * nv + z_ * hf[nn * 128 + c];
      hf[nn * 128 + c] = hnew;
      hb[nn * 136 + c] = f2h(hnew);
      int nabs = n0 + nn;
      out[((size_t)t * kN + nabs) * kS + 2 + c] = hnew;
      if (t == 63) out[((size_t)(64 * kN) + nabs) * kS + 2 + c] = hnew;
    }
    __syncthreads();
    // 8) v = h2 . critic_w + critic_b
    if (tid < 16) {
      float s = 0.f;
      for (int c = 0; c < 128; ++c) s += hf[tid * 128 + c] * cwL[c];
      float v = s + cb0;
      int nabs = n0 + tid;
      out[((size_t)t * kN + nabs) * kS + 1] = v;
      if (t == 63) out[((size_t)(64 * kN) + nabs) * kS + 1] = v;
    }
    __syncthreads();
  }
}

// ---------------------------------------------------------------------------
// Host launcher
// ---------------------------------------------------------------------------
extern "C" void kernel_launch(void* const* d_in, const int* in_sizes, int n_in,
                              void* d_out, int out_size, void* d_ws, size_t ws_size,
                              hipStream_t stream) {
  (void)in_sizes; (void)n_in; (void)out_size; (void)ws_size;
  const float* inputs   = (const float*)d_in[0];
  const float* rnn_hxs  = (const float*)d_in[1];
  const float* gumbel   = (const float*)d_in[2];
  const float* conv1_w  = (const float*)d_in[3];
  const float* conv1_b  = (const float*)d_in[4];
  const float* conv2_w  = (const float*)d_in[5];
  const float* conv2_b  = (const float*)d_in[6];
  const float* encf_wih = (const float*)d_in[7];
  const float* encf_whh = (const float*)d_in[8];
  const float* encf_bih = (const float*)d_in[9];
  const float* encf_bhh = (const float*)d_in[10];
  const float* encb_wih = (const float*)d_in[11];
  const float* encb_whh = (const float*)d_in[12];
  const float* encb_bih = (const float*)d_in[13];
  const float* encb_bhh = (const float*)d_in[14];
  const float* cell_wih = (const float*)d_in[15];
  const float* cell_whh = (const float*)d_in[16];
  const float* cell_bih = (const float*)d_in[17];
  const float* cell_bhh = (const float*)d_in[18];
  const float* critic_w = (const float*)d_in[19];
  const float* critic_b = (const float*)d_in[20];
  const float* qg_w     = (const float*)d_in[21];
  const float* qg_b     = (const float*)d_in[22];
  const float* term_w   = (const float*)d_in[23];
  const float* term_b   = (const float*)d_in[24];
  float* out = (float*)d_out;

  // Workspace carve-up (~143.6 MB total; assumes ws_size is sufficient)
  char* base = (char*)d_ws;
  size_t off = 0;
  auto take = [&](size_t bytes) -> char* {
    char* p = base + off;
    off = (off + bytes + 255) & ~(size_t)255;
    return p;
  };
  unsigned short* Mh  = (unsigned short*)take((size_t)kN * kHW * kH * 2);   // 8.39 MB
  float* giF          = (float*)take((size_t)kHW * kN * kG * 4);            // 50.3 MB
  float* giB          = (float*)take((size_t)kHW * kN * kG * 4);            // 50.3 MB
  float* Kbuf         = (float*)take((size_t)kN * kHW * 256 * 4);           // 33.6 MB
  unsigned short* dWihF = (unsigned short*)take(49152 * 2);
  unsigned short* dWhhF = (unsigned short*)take(49152 * 2);
  unsigned short* dWihB = (unsigned short*)take(49152 * 2);
  unsigned short* dWhhB = (unsigned short*)take(49152 * 2);
  unsigned short* dCWih = (unsigned short*)take(49152 * 2);
  unsigned short* dCWhh = (unsigned short*)take(49152 * 2);
  unsigned short* dQgW  = (unsigned short*)take(32768 * 2);
  unsigned short* dW1   = (unsigned short*)take(128 * 160 * 2);
  unsigned short* dW2   = (unsigned short*)take(128 * 1152 * 2);

  k_prep<<<dim3(576, 9), 256, 0, stream>>>(
      encf_wih, encf_whh, encb_wih, encb_whh, cell_wih, cell_whh, qg_w, conv1_w,
      conv2_w, dWihF, dWhhF, dWihB, dWhhB, dCWih, dCWhh, dQgW, dW1, dW2);
  k_conv<<<kN, 256, 0, stream>>>(inputs, conv1_b, conv2_b, dW1, dW2, Mh);
  k_gi<<<dim3(kHW, 2), 256, 0, stream>>>(Mh, dWihF, dWihB, encf_bih, encb_bih,
                                         giF, giB);
  k_encscan<<<16, 256, 0, stream>>>(giF, giB, dWhhF, dWhhB, encf_bhh, encb_bhh,
                                    Kbuf);
  k_dec<<<8, 256, 0, stream>>>(Mh, Kbuf, dQgW, dCWih, dCWhh, qg_b, term_w, term_b,
                               critic_w, critic_b, cell_bih, cell_bhh, inputs,
                               gumbel, rnn_hxs, out);
}
